// Generator_55250459296107
// MI455X (gfx1250) — matrix operand order repaired and checked by reading, not compiled, not run on hardware
//
#include <hip/hip_runtime.h>
#include <cmath>

// ---------------------------------------------------------------------------
// Types for CDNA5 WMMA (wave32, 16x16x32 bf16 -> f32)
// ---------------------------------------------------------------------------
typedef __attribute__((ext_vector_type(16))) __bf16        v16bf;
typedef __attribute__((ext_vector_type(8)))  float         v8f;
typedef __attribute__((ext_vector_type(16))) unsigned short v16u;
typedef __attribute__((ext_vector_type(8)))  unsigned short v8u;

__device__ __forceinline__ unsigned short f2bf(float f) {
    union { float f; unsigned int u; } x; x.f = f;
    unsigned int r = x.u + 0x7FFFu + ((x.u >> 16) & 1u);   // RNE
    return (unsigned short)(r >> 16);
}
__device__ __forceinline__ float bf2f(unsigned short h) {
    union { unsigned int u; float f; } x; x.u = ((unsigned int)h) << 16;
    return x.f;
}

#define SQRT2F 1.41421356237309515f
#define LRELU(v) (((v) > 0.f ? (v) : 0.2f * (v)) * SQRT2F)

// ---------------------------------------------------------------------------
// Mapping network
// ---------------------------------------------------------------------------
__global__ __launch_bounds__(512) void pixelnorm_kernel(const float* __restrict__ z,
                                                        float* __restrict__ out) {
    const int b = blockIdx.x, j = threadIdx.x;
    __shared__ float red[512];
    float v = z[b * 512 + j];
    red[j] = v * v;
    __syncthreads();
    for (int st = 256; st > 0; st >>= 1) {
        if (j < st) red[j] += red[j + st];
        __syncthreads();
    }
    out[b * 512 + j] = v * rsqrtf(red[0] / 512.f + 1e-8f);
}

__global__ __launch_bounds__(256) void map_linear_kernel(const float* __restrict__ xin,
                                                         const float* __restrict__ w,
                                                         const float* __restrict__ bias,
                                                         float* __restrict__ xout) {
    const int idx = blockIdx.x * 256 + threadIdx.x;
    if (idx >= 4 * 512) return;
    const int b = idx >> 9, o = idx & 511;
    const float scale = 0.01f * 0.044194173824159216f;  // lr_mul / sqrt(512)
    const float* xr = xin + b * 512;
    const float* wr = w + (size_t)o * 512;
    float acc = 0.f;
    for (int j = 0; j < 512; ++j) acc += xr[j] * wr[j];
    float v = acc * scale + bias[o] * 0.01f;
    xout[idx] = LRELU(v);
}

// ---------------------------------------------------------------------------
// Style projection:  s[b,i] = mod_b[i] + (1/sqrt(512)) * <wlat[b,:], mod_w[i,:]>
// ---------------------------------------------------------------------------
__global__ __launch_bounds__(256) void style_kernel(const float* __restrict__ wlat,
                                                    const float* __restrict__ mod_w,
                                                    const float* __restrict__ mod_b,
                                                    float* __restrict__ s, int I) {
    const int idx = blockIdx.x * 256 + threadIdx.x;
    if (idx >= 4 * I) return;
    const int b = idx / I, i = idx % I;
    const float* wl = wlat + b * 512;
    const float* mw = mod_w + (size_t)i * 512;
    float acc = 0.f;
    for (int j = 0; j < 512; ++j) acc += wl[j] * mw[j];
    s[idx] = mod_b[i] + acc * 0.044194173824159216f;
}

// ---------------------------------------------------------------------------
// Modulate + demodulate a 3x3 conv weight -> bf16 [b][O][tap][I]
// ---------------------------------------------------------------------------
__global__ __launch_bounds__(256) void modulate_conv_kernel(const float* __restrict__ weight,
                                                            const float* __restrict__ s,
                                                            unsigned short* __restrict__ wout,
                                                            int I, int O) {
    const int o = blockIdx.x, b = blockIdx.y;
    const float scale = rsqrtf((float)(I * 9));
    const float* wrow = weight + (size_t)o * I * 9;
    const float* sb = s + b * I;
    __shared__ float red[256];
    float loc = 0.f;
    for (int idx = threadIdx.x; idx < I * 9; idx += 256) {
        const int i = idx / 9, t = idx % 9;
        float w = scale * wrow[(size_t)i * 9 + t] * sb[i];
        loc += w * w;
    }
    red[threadIdx.x] = loc;
    __syncthreads();
    for (int st = 128; st > 0; st >>= 1) {
        if (threadIdx.x < st) red[threadIdx.x] += red[threadIdx.x + st];
        __syncthreads();
    }
    const float d = rsqrtf(red[0] + 1e-8f);
    unsigned short* out = wout + ((size_t)b * O + o) * 9 * I;
    for (int idx = threadIdx.x; idx < I * 9; idx += 256) {
        const int i = idx / 9, t = idx % 9;
        float w = scale * wrow[(size_t)i * 9 + t] * sb[i] * d;
        out[(size_t)t * I + i] = f2bf(w);
    }
}

// ---------------------------------------------------------------------------
// Upsample conv: fold conv_transpose(stride2) + 4-tap blur into 4 polyphase
// 3x3 kernels.  K6[t][s] = sum_{p,q} b1[p] b1[q] w3[t+p-3][s+q-3].
// Conv kernel gathers in[yy + dy - 1]; tap (dy,dx) corresponds to ty=2-dy,
// so we store out[tap(dy,dx)] = K6[2*(2-dy)+py][2*(2-dx)+px].
// Layout: [ (b*4+phase) ][O][tap][I], phase = py*2+px.
// ---------------------------------------------------------------------------
__global__ __launch_bounds__(256) void modulate_up_kernel(const float* __restrict__ weight,
                                                          const float* __restrict__ s,
                                                          unsigned short* __restrict__ wout,
                                                          int I, int O) {
    const int o = blockIdx.x, b = blockIdx.y;
    const float scale = rsqrtf((float)(I * 9));
    const float* wrow = weight + (size_t)o * I * 9;
    const float* sb = s + b * I;
    __shared__ float red[256];
    float loc = 0.f;
    for (int i = threadIdx.x; i < I; i += 256) {
        const float si = sb[i] * scale;
        #pragma unroll
        for (int t = 0; t < 9; ++t) {
            float w = wrow[(size_t)i * 9 + t] * si;
            loc += w * w;
        }
    }
    red[threadIdx.x] = loc;
    __syncthreads();
    for (int st = 128; st > 0; st >>= 1) {
        if (threadIdx.x < st) red[threadIdx.x] += red[threadIdx.x + st];
        __syncthreads();
    }
    const float d = rsqrtf(red[0] + 1e-8f);
    const float b1[4] = {0.25f, 0.75f, 0.75f, 0.25f};
    for (int i = threadIdx.x; i < I; i += 256) {
        const float si = sb[i] * scale * d;
        float w3[3][3];
        #pragma unroll
        for (int dy = 0; dy < 3; ++dy)
            #pragma unroll
            for (int dx = 0; dx < 3; ++dx)
                w3[dy][dx] = wrow[(size_t)i * 9 + dy * 3 + dx] * si;
        float K6[6][6];
        #pragma unroll
        for (int t = 0; t < 6; ++t)
            #pragma unroll
            for (int ss = 0; ss < 6; ++ss) {
                float acc = 0.f;
                #pragma unroll
                for (int p = 0; p < 4; ++p) {
                    const int r = t + p - 3;
                    if (r < 0 || r > 2) continue;
                    #pragma unroll
                    for (int q = 0; q < 4; ++q) {
                        const int c = ss + q - 3;
                        if (c < 0 || c > 2) continue;
                        acc += b1[p] * b1[q] * w3[r][c];
                    }
                }
                K6[t][ss] = acc;
            }
        #pragma unroll
        for (int ph = 0; ph < 4; ++ph) {
            const int py = ph >> 1, px = ph & 1;
            unsigned short* out = wout + ((size_t)(b * 4 + ph) * O + o) * 9 * I;
            #pragma unroll
            for (int dy = 0; dy < 3; ++dy)
                #pragma unroll
                for (int dx = 0; dx < 3; ++dx)
                    out[(size_t)(dy * 3 + dx) * I + i] =
                        f2bf(K6[2 * (2 - dy) + py][2 * (2 - dx) + px]);
        }
    }
}

// ---------------------------------------------------------------------------
// Implicit-GEMM 3x3 conv, NHWC bf16 activations, WMMA 16x16x32 bf16.
// Per workgroup: one batch, 64 out-channels, one 8x8 pixel tile at gather
// resolution HxW.  The 10x10 halo for each 32-channel chunk is streamed to
// LDS with GLOBAL_LOAD_ASYNC_TO_LDS_B128 (ASYNCcnt), double-buffered: chunk
// c+1 is in flight while the 9 taps of chunk c run from LDS.  OOB halo slots
// are zero-filled once and never overwritten (invalid lanes redirect to a
// dummy LDS slot so every wave issues exactly 2 async ops per chunk).
// up=1 scatters output to (2y+py, 2x+px) with phase-specific weights.
// Epilogue fuses noise + bias + leaky_relu*sqrt(2); 8 consecutive channels
// per lane -> one packed 16B store.
// ---------------------------------------------------------------------------
#define HSTRIDE 40                 // halo pixel stride in halves (80B)
#define HBUF    (100 * HSTRIDE)    // one halo buffer: 4000 halves = 8000B

__global__ __launch_bounds__(256) void conv3x3_wmma_kernel(
        const unsigned short* __restrict__ act,    // [B][H][W][I] bf16
        const unsigned short* __restrict__ wmod,   // [B(*4)][O][9][I] bf16
        const float* __restrict__ noise,           // [Ho][Wo]
        const float* __restrict__ act_bias,        // [O]
        const float* __restrict__ noise_wp,        // scalar
        unsigned short* __restrict__ outact,       // [B][Ho][Wo][O] bf16
        int I, int O, int H, int W, int up) {
    const int b = blockIdx.z;
    const int nPhase = up ? 4 : 1;
    const int tw = (W + 7) >> 3;
    int bx = blockIdx.x;
    const int phase = bx % nPhase; bx /= nPhase;
    const int tX = bx % tw, tY = bx / tw;
    const int oBase = blockIdx.y * 64;
    const int py = phase >> 1, px = phase & 1;

    const int tid = threadIdx.x;
    const int lane = tid & 31, wave = tid >> 5;
    const int wm = wave & 3, wn = wave >> 2;

    __shared__ unsigned short sH[2 * HBUF + 8];    // double buffer + dummy slot

    v8f acc0 = {};
    v8f acc1 = {};

    const unsigned short* wp =
        wmod + (size_t)(up ? (b * 4 + phase) : b) * O * 9 * I;

    const int n16  = lane & 15;
    const int kb   = (lane >> 4) * 16;   // A: K base within chunk {0,16}
    const int koff = (lane >> 4) * 16;   // B: K base within chunk {0,16}
    const int om   = oBase + wm * 16 + n16;
    const unsigned short* aptr = wp + (size_t)om * 9 * I;

    // B-fragment halo-pixel base (per lane): pixel p0 = wn*32 + n16
    const int p0    = wn * 32 + n16;
    const int hbase = ((p0 >> 3) * 10 + (p0 & 7)) * HSTRIDE + koff;

    // Halo staging precompute: 100 pixels x 4 x 16B units = 400 units
    int    dst_off[2];
    size_t src_off[2];
    bool   okf[2];
    #pragma unroll
    for (int t = 0; t < 2; ++t) {
        const int u = tid + t * 256;
        const int hp = u >> 2, part = u & 3;
        const int hy = hp / 10, hx = hp % 10;
        const int gy = tY * 8 + hy - 1, gx = tX * 8 + hx - 1;
        okf[t] = (u < 400) && gy >= 0 && gy < H && gx >= 0 && gx < W;
        dst_off[t] = hp * HSTRIDE + part * 8;
        src_off[t] = ((size_t)(b * H + gy) * W + gx) * I + part * 8;
    }

    // Zero both halo buffers (OOB slots stay zero; valid slots overwritten)
    for (int k = tid * 8; k < 2 * HBUF; k += 256 * 8) *(v8u*)&sH[k] = (v8u){};
    __syncthreads();

    const unsigned sbase  = (unsigned)(size_t)&sH[0];
    const unsigned dummyA = sbase + 2 * HBUF * 2;   // byte address of dummy

    // Issue one chunk's halo as exactly-two async LDS loads per thread.
    auto issue = [&](int c) {
        const unsigned bs = (unsigned)((c & 1) * HBUF * 2);
        const unsigned d0 = okf[0] ? (sbase + bs + (unsigned)dst_off[0] * 2) : dummyA;
        const unsigned d1 = okf[1] ? (sbase + bs + (unsigned)dst_off[1] * 2) : dummyA;
        const unsigned short* g0 = okf[0] ? (act + src_off[0] + c * 32) : act;
        const unsigned short* g1 = okf[1] ? (act + src_off[1] + c * 32) : act;
        asm volatile(
            "global_load_async_to_lds_b128 %0, %2, off\n\t"
            "global_load_async_to_lds_b128 %1, %3, off\n\t"
            "s_wait_xcnt 0x0"
            :: "v"(d0), "v"(d1), "v"(g0), "v"(g1)
            : "memory");
    };

    const int NC = I >> 5;   // 32-channel chunks
    issue(0);
    for (int c = 0; c < NC; ++c) {
        if (c + 1 < NC) {
            issue(c + 1);                                    // prefetch next
            asm volatile("s_wait_asynccnt 2" ::: "memory");  // chunk c landed
        } else {
            asm volatile("s_wait_asynccnt 0" ::: "memory");
        }
        __syncthreads();

        const unsigned short* sb = sH + (c & 1) * HBUF;
        const int i0 = c * 32;
        #pragma unroll
        for (int tap = 0; tap < 9; ++tap) {
            const int dy = tap / 3, dx = tap % 3;
            // A fragment (contiguous K: lane<16 -> K{0..15}, lane>=16 -> K{16..31})
            const unsigned short* arow = aptr + (size_t)tap * I + i0;
            v8u alo = *(const v8u*)(arow + kb);
            v8u ahi = *(const v8u*)(arow + kb + 8);
            v16u av = __builtin_shufflevector(alo, ahi, 0, 1, 2, 3, 4, 5, 6, 7,
                                              8, 9, 10, 11, 12, 13, 14, 15);
            v16bf a = __builtin_bit_cast(v16bf, av);

            // B fragments from LDS halo (per-lane 16 contiguous channels)
            const int h0 = hbase + (dy * 10 + dx) * HSTRIDE;
            v8u bl0 = *(const v8u*)&sb[h0];
            v8u bh0 = *(const v8u*)&sb[h0 + 8];
            v16u bv0 = __builtin_shufflevector(bl0, bh0, 0, 1, 2, 3, 4, 5, 6, 7,
                                               8, 9, 10, 11, 12, 13, 14, 15);
            v8u bl1 = *(const v8u*)&sb[h0 + 20 * HSTRIDE];
            v8u bh1 = *(const v8u*)&sb[h0 + 20 * HSTRIDE + 8];
            v16u bv1 = __builtin_shufflevector(bl1, bh1, 0, 1, 2, 3, 4, 5, 6, 7,
                                               8, 9, 10, 11, 12, 13, 14, 15);

            acc0 = __builtin_amdgcn_wmma_f32_16x16x32_bf16(
                false, a, false, __builtin_bit_cast(v16bf, bv0),
                (short)0, acc0, false, false);
            acc1 = __builtin_amdgcn_wmma_f32_16x16x32_bf16(
                false, a, false, __builtin_bit_cast(v16bf, bv1),
                (short)0, acc1, false, false);
        }
        __syncthreads();   // buffer (c&1) free before chunk c+2 streams in
    }

    // Epilogue: noise + bias + fused leaky relu, packed NHWC store
    const float nw = noise_wp[0];
    const int half = lane >> 4;
    const int Ho = up ? H * 2 : H, Wo = up ? W * 2 : W;
    const int oCh = oBase + wm * 16 + half * 8;   // 8 consecutive channels
    #pragma unroll
    for (int t = 0; t < 2; ++t) {
        v8f acc = t ? acc1 : acc0;
        const int p = wn * 32 + t * 16 + n16;
        const int pr = p >> 3, pc = p & 7;
        const int y = tY * 8 + pr, x = tX * 8 + pc;
        if (y < H && x < W) {
            const int oy = up ? 2 * y + py : y;
            const int ox = up ? 2 * x + px : x;
            const float nz = nw * noise[(size_t)oy * Wo + ox];
            v8u pk;
            #pragma unroll
            for (int r = 0; r < 8; ++r) {
                float v = acc[r] + nz + act_bias[oCh + r];
                pk[r] = f2bf(LRELU(v));
            }
            *(v8u*)(outact + ((size_t)(b * Ho + oy) * Wo + ox) * O + oCh) = pk;
        }
    }
}

// ---------------------------------------------------------------------------
// to_rgb: 1x1 modulated conv (no demod) weights [B][3][I]
// ---------------------------------------------------------------------------
__global__ __launch_bounds__(256) void modulate_rgb_kernel(const float* __restrict__ weight,
                                                           const float* __restrict__ s,
                                                           float* __restrict__ wout, int I) {
    const int idx = blockIdx.x * 256 + threadIdx.x;
    if (idx >= 4 * I) return;
    const int b = idx / I, i = idx % I;
    const float si = s[idx] * rsqrtf((float)I);
    #pragma unroll
    for (int c = 0; c < 3; ++c)
        wout[((size_t)b * 3 + c) * I + i] = weight[(size_t)c * I + i] * si;
}

// 1x1 conv to RGB (NHWC act, contiguous I) + bias + fused 4-tap polyphase
// upsample of the NCHW skip.  Output NCHW (matches reference layout).
__global__ __launch_bounds__(256) void to_rgb_kernel(const unsigned short* __restrict__ act,
                                                     const float* __restrict__ wrgb,
                                                     const float* __restrict__ bias,
                                                     const float* __restrict__ skipIn,
                                                     float* __restrict__ out,
                                                     int I, int H, int W) {
    const int idx = blockIdx.x * 256 + threadIdx.x;
    const int HW = H * W;
    if (idx >= 4 * HW) return;
    const int b = idx / HW, p = idx % HW;
    const unsigned short* ab = act + (size_t)(b * HW + p) * I;
    const float* w0 = wrgb + (size_t)b * 3 * I;
    float a0 = 0.f, a1 = 0.f, a2 = 0.f;
    for (int i = 0; i < I; ++i) {
        const float v = bf2f(ab[i]);
        a0 += w0[i] * v;
        a1 += w0[I + i] * v;
        a2 += w0[2 * I + i] * v;
    }
    const int y = p / W, x = p % W;
    float sk[3] = {0.f, 0.f, 0.f};
    if (skipIn) {
        const float b1[4] = {0.25f, 0.75f, 0.75f, 0.25f};
        const int Hs = H >> 1, Ws = W >> 1;
        const int pyy = y & 1, pxx = x & 1;
        const int my0 = (y - 2 + pyy) >> 1;
        const int mx0 = (x - 2 + pxx) >> 1;
        const float cy[2] = {b1[pyy], b1[2 + pyy]};
        const float cx[2] = {b1[pxx], b1[2 + pxx]};
        #pragma unroll
        for (int jy = 0; jy < 2; ++jy) {
            const int my = my0 + jy;
            if (my < 0 || my >= Hs) continue;
            #pragma unroll
            for (int jx = 0; jx < 2; ++jx) {
                const int mx = mx0 + jx;
                if (mx < 0 || mx >= Ws) continue;
                const float cf = cy[jy] * cx[jx];
                #pragma unroll
                for (int c = 0; c < 3; ++c)
                    sk[c] += cf * skipIn[(((size_t)b * 3 + c) * Hs + my) * Ws + mx];
            }
        }
    }
    const float acc[3] = {a0, a1, a2};
    #pragma unroll
    for (int c = 0; c < 3; ++c)
        out[(((size_t)b * 3 + c) * H + y) * W + x] = acc[c] + bias[c] + sk[c];
}

// ConstantInput: broadcast const [512][4][4] over batch, f32 -> bf16 NHWC
__global__ __launch_bounds__(256) void const_init_kernel(const float* __restrict__ cst,
                                                         unsigned short* __restrict__ act) {
    const int idx = blockIdx.x * 256 + threadIdx.x;
    if (idx >= 4 * 16 * 512) return;
    const int r = idx & 8191;         // within one batch image
    const int p = r >> 9, c = r & 511;
    act[idx] = f2bf(cst[c * 16 + p]);
}

// ---------------------------------------------------------------------------
// Workspace arena (bytes)
// ---------------------------------------------------------------------------
static const size_t OFF_T0   = 0;              // mapping ping (final latent w)
static const size_t OFF_T1   = 8 * 1024;       // mapping pong
static const size_t OFF_S    = 16 * 1024;      // style s[b][I]
static const size_t OFF_WRGB = 24 * 1024;      // rgb weights [B][3][I] f32
static const size_t OFF_WMOD = 64 * 1024;      // modulated bf16 weights (<=75.5MB)
static const size_t OFF_ACT0 = (size_t)40 << 20;
static const size_t OFF_ACT1 = (size_t)76 << 20;
static const size_t OFF_SK0  = (size_t)112 << 20;
static const size_t OFF_SK1  = OFF_SK0 + ((size_t)1 << 20);

extern "C" void kernel_launch(void* const* d_in, const int* in_sizes, int n_in,
                              void* d_out, int out_size, void* d_ws, size_t ws_size,
                              hipStream_t stream) {
    (void)in_sizes; (void)n_in; (void)out_size; (void)ws_size;
    char* ws = (char*)d_ws;
    float* t0 = (float*)(ws + OFF_T0);
    float* t1 = (float*)(ws + OFF_T1);
    float* sbuf = (float*)(ws + OFF_S);
    float* wrgb = (float*)(ws + OFF_WRGB);
    unsigned short* wmod = (unsigned short*)(ws + OFF_WMOD);
    unsigned short* act0 = (unsigned short*)(ws + OFF_ACT0);
    unsigned short* act1 = (unsigned short*)(ws + OFF_ACT1);
    float* sk0 = (float*)(ws + OFF_SK0);
    float* sk1 = (float*)(ws + OFF_SK1);

    // ---- mapping network: PixelNorm + 8x EqualLinear(lrelu) ----
    pixelnorm_kernel<<<4, 512, 0, stream>>>((const float*)d_in[0], t0);
    float* bufs[2] = {t0, t1};
    for (int l = 0; l < 8; ++l)
        map_linear_kernel<<<8, 256, 0, stream>>>(bufs[l & 1],
                                                 (const float*)d_in[1 + 2 * l],
                                                 (const float*)d_in[2 + 2 * l],
                                                 bufs[(l + 1) & 1]);
    const float* wlat = bufs[0];  // layer 7 wrote t0

    // ---- constant input ----
    const_init_kernel<<<(4 * 16 * 512 + 255) / 256, 256, 0, stream>>>(
        (const float*)d_in[17], act0);

    auto run_styled = [&](int pbase, const unsigned short* inAct,
                          unsigned short* outAct, int I, int O, int H, int up,
                          int noiseIdx) {
        style_kernel<<<(4 * I + 255) / 256, 256, 0, stream>>>(
            wlat, (const float*)d_in[pbase + 1], (const float*)d_in[pbase + 2],
            sbuf, I);
        if (up)
            modulate_up_kernel<<<dim3(O, 4), 256, 0, stream>>>(
                (const float*)d_in[pbase], sbuf, wmod, I, O);
        else
            modulate_conv_kernel<<<dim3(O, 4), 256, 0, stream>>>(
                (const float*)d_in[pbase], sbuf, wmod, I, O);
        const int tw = (H + 7) / 8;
        const int nP = up ? 4 : 1;
        dim3 g(tw * tw * nP, O / 64, 4);
        conv3x3_wmma_kernel<<<g, 256, 0, stream>>>(
            inAct, wmod, (const float*)d_in[noiseIdx],
            (const float*)d_in[pbase + 4], (const float*)d_in[pbase + 3],
            outAct, I, O, H, H, up);
    };
    auto run_rgb = [&](int pbase, const unsigned short* inAct,
                       const float* skipIn, float* skipOut, int I, int H) {
        style_kernel<<<(4 * I + 255) / 256, 256, 0, stream>>>(
            wlat, (const float*)d_in[pbase + 1], (const float*)d_in[pbase + 2],
            sbuf, I);
        modulate_rgb_kernel<<<(4 * I + 255) / 256, 256, 0, stream>>>(
            (const float*)d_in[pbase], sbuf, wrgb, I);
        to_rgb_kernel<<<(4 * H * H + 255) / 256, 256, 0, stream>>>(
            inAct, wrgb, (const float*)d_in[pbase + 3], skipIn, skipOut, I, H, H);
    };

    // conv1 (4x4, 512->512) + to_rgb1
    run_styled(18, act0, act1, 512, 512, 4, 0, 97);
    run_rgb(23, act1, nullptr, sk0, 512, 4);

    unsigned short* cur = act1;
    unsigned short* alt = act0;
    float* prevSkip = sk0;
    int res = 4, in_c = 512;
    for (int t = 0; t < 5; ++t) {
        const int out_c = (t == 4) ? 256 : 512;
        const int pconv = 27 + 10 * t;
        run_styled(pconv, cur, alt, in_c, out_c, res, 1, 97 + 1 + 2 * t);       // upsample
        run_styled(pconv + 5, alt, cur, out_c, out_c, res * 2, 0, 97 + 2 + 2 * t);
        res *= 2;
        float* dst = (t == 4) ? (float*)d_out : ((t & 1) ? sk0 : sk1);
        run_rgb(77 + 4 * t, cur, prevSkip, dst, out_c, res);
        prevSkip = dst;
        in_c = out_c;
    }
}
